// LIIF_9405978378848
// MI455X (gfx1250) — matrix operand order repaired and checked by reading, not compiled, not run on hardware
//
#include <hip/hip_runtime.h>

// ---------------------------------------------------------------------------
// LIIF fused inference for gfx1250 (MI455X).
//   conv3x3  -> f32 feature map (workspace)
//   pack     -> bf16 weights pre-swizzled into CDNA5 WMMA B-fragment layout
//   fused    -> per-wave 32-query MLP via v_wmma_f32_16x16x32_bf16
//               (two 16-row A tiles share every B fragment -> 2x L2 reuse),
//               4 local-ensemble shifts + bilinear skip, f32 out.
// Workspace use: 2 MB feat + ~0.72 MB packed weights (~3 MB total).
// ---------------------------------------------------------------------------

typedef unsigned int  u32;
typedef unsigned short u16;
typedef __attribute__((ext_vector_type(16))) __bf16 v16bf;
typedef __attribute__((ext_vector_type(8)))  float  v8f;
typedef __attribute__((ext_vector_type(4)))  u32    v4u;

static constexpr int BATCH = 2;
static constexpr int CH    = 64;
static constexpr int H     = 64;
static constexpr int W     = 64;
static constexpr int Q     = 192 * 192;      // 36864
static constexpr int ROWS  = BATCH * Q;      // 73728
static constexpr int MROW  = 32;             // queries per wave
static constexpr int TILES = ROWS / MROW;    // 2304
static constexpr int K0P   = 640;            // 624 padded to 32
static constexpr int KT0   = K0P / 32;       // 20
static constexpr int HID   = 256;
static constexpr int KTH   = HID / 32;       // 8

// workspace layout (bytes)
static constexpr size_t FEAT_BYTES = (size_t)BATCH * CH * H * W * 4;   // 2 MB
static constexpr size_t P0_BYTES   = (size_t)16 * KT0 * 256 * 4;       // 320 KB
static constexpr size_t PH_BYTES   = (size_t)16 * KTH * 256 * 4;       // 128 KB

#define DEVFN __device__ __forceinline__

DEVFN u16 f2bf(float f) {                      // round-to-nearest-even f32->bf16
    u32 u = __float_as_uint(f);
    u32 r = (u + 0x7FFFu + ((u >> 16) & 1u)) >> 16;
    return (u16)r;
}

union ABReg { u32 u[8]; v4u q[2]; v16bf v; };

// A fragment (16x32 bf16, ISA 7.12.2 A-layout) from a row-major LDS panel.
// lane L: row = rowBase + (L&15); VGPR j holds
//   K = kbase + (L>>4)*8 + (j>>2)*16 + (j&3)*2 (+1)
DEVFN v16bf load_a_frag(const u32* panel, int rowDw, int lane, int kbase, int rowBase) {
    const int r  = rowBase + (lane & 15);
    const int hh = lane >> 4;
    ABReg t;
#pragma unroll
    for (int j = 0; j < 8; ++j) {
        int k = kbase + hh * 8 + (j >> 2) * 16 + (j & 3) * 2;
        t.u[j] = panel[r * rowDw + (k >> 1)];
    }
    return t.v;
}

// B fragment (32x16 bf16) from packed weights: per fragment 32 lanes x 8 dwords,
// per-lane contiguous -> two b128 loads.
DEVFN v16bf load_b_frag(const v4u* __restrict__ pack, int frag, int lane) {
    ABReg t;
    const v4u* p = pack + ((size_t)frag * 32 + lane) * 2;
    t.q[0] = p[0];
    t.q[1] = p[1];
    return t.v;
}

// One hidden layer for 32 rows: [32 x Kp](LDS bf16) x [Kp x 256] -> ReLU -> outPanel.
// Two 16-row A tiles share every B fragment (halves L2 weight traffic per WMMA).
DEVFN void run_layer32(const u32* apanel, int rowDw, int KT,
                       const v4u* __restrict__ pack, const float* __restrict__ bias,
                       u16* outPanel, int lane) {
    const int col = lane & 15;
    const int hh  = lane >> 4;
    for (int ntg = 0; ntg < 16; ntg += 4) {
        v8f acc[2][4];
#pragma unroll
        for (int m = 0; m < 2; ++m)
#pragma unroll
            for (int t = 0; t < 4; ++t) acc[m][t] = (v8f){0, 0, 0, 0, 0, 0, 0, 0};
        for (int kt = 0; kt < KT; ++kt) {
            v16bf a0 = load_a_frag(apanel, rowDw, lane, kt * 32, 0);
            v16bf a1 = load_a_frag(apanel, rowDw, lane, kt * 32, 16);
#pragma unroll
            for (int t = 0; t < 4; ++t) {
                v16bf bb = load_b_frag(pack, (ntg + t) * KT + kt, lane);
                acc[0][t] = __builtin_amdgcn_wmma_f32_16x16x32_bf16(
                    false, a0, false, bb, (short)0, acc[0][t], false, false);
                acc[1][t] = __builtin_amdgcn_wmma_f32_16x16x32_bf16(
                    false, a1, false, bb, (short)0, acc[1][t], false, false);
            }
        }
#pragma unroll
        for (int t = 0; t < 4; ++t) {
            int   n  = (ntg + t) * 16 + col;
            float bi = bias[n];
#pragma unroll
            for (int m = 0; m < 2; ++m)
#pragma unroll
                for (int v = 0; v < 8; ++v) {
                    float f = fmaxf(acc[m][t][v] + bi, 0.f);
                    outPanel[(m * 16 + v + 8 * hh) * HID + n] = f2bf(f);
                }
        }
    }
}

// ---------------------------------------------------------------------------
__global__ void liif_conv3x3_kernel(const float* __restrict__ inp,
                                    const float* __restrict__ cw,
                                    const float* __restrict__ cb,
                                    float* __restrict__ feat) {
    int idx = blockIdx.x * blockDim.x + threadIdx.x;          // ((b*64+co)*H+y)*W+x
    if (idx >= BATCH * CH * H * W) return;
    int x  = idx % W;
    int y  = (idx / W) % H;
    int co = (idx / (W * H)) % CH;
    int b  = idx / (W * H * CH);
    float s = cb[co];
#pragma unroll
    for (int ci = 0; ci < 3; ++ci) {
        const float* im = inp + ((size_t)(b * 3 + ci)) * H * W;
        const float* wk = cw + ((size_t)(co * 3 + ci)) * 9;
#pragma unroll
        for (int t = 0; t < 9; ++t) {
            int yy = y + t / 3 - 1, xx = x + t % 3 - 1;
            if (yy >= 0 && yy < H && xx >= 0 && xx < W)
                s += im[yy * W + xx] * wk[t];
        }
    }
    feat[idx] = s;
}

// Pre-swizzle a row-major f32 weight [Ksrc x Nsrc] into bf16 WMMA-B fragments.
__global__ void liif_pack_kernel(const float* __restrict__ w, u32* __restrict__ dst,
                                 int Ksrc, int Nsrc, int KT, int NT) {
    int idx   = blockIdx.x * blockDim.x + threadIdx.x;
    int total = NT * KT * 256;
    if (idx >= total) return;
    int j    = idx & 7;
    int lane = (idx >> 3) & 31;
    int frag = idx >> 8;
    int nt   = frag / KT;
    int kt   = frag - nt * KT;
    int n    = nt * 16 + (lane & 15);
    int k    = kt * 32 + (lane >> 4) * 8 + (j >> 2) * 16 + (j & 3) * 2;
    float f0 = (k < Ksrc && n < Nsrc) ? w[(size_t)k * Nsrc + n] : 0.f;
    float f1 = (k + 1 < Ksrc && n < Nsrc) ? w[(size_t)(k + 1) * Nsrc + n] : 0.f;
    dst[idx] = (u32)f2bf(f0) | ((u32)f2bf(f1) << 16);
}

// ---------------------------------------------------------------------------
__global__ void __launch_bounds__(32)
liif_fused_kernel(const float* __restrict__ feat, const float* __restrict__ inp,
                  const float* __restrict__ coord, const float* __restrict__ cell,
                  const v4u* __restrict__ p0, const v4u* __restrict__ p1,
                  const v4u* __restrict__ p2, const v4u* __restrict__ p3,
                  const v4u* __restrict__ p4,
                  const float* __restrict__ b0, const float* __restrict__ b1,
                  const float* __restrict__ b2, const float* __restrict__ b3,
                  const float* __restrict__ b4,
                  float* __restrict__ out) {
    // LDS arena: xpanel [32 x 640] bf16 (40 KB) ; hA [32 x 256] bf16 (16 KB).
    // hB aliases the (dead after layer-0) xpanel region: total ~57 KB/WG.
    __shared__ u16  smem[MROW * K0P + MROW * HID];
    __shared__ float wgt_s[4][MROW];
    __shared__ float bil_s[MROW][3];
    u16* xpanel = smem;               // 32*640
    u16* hB     = smem;               // 32*256, aliases xpanel (phase-disjoint)
    u16* hA     = smem + MROW * K0P;  // 32*256

    const int  lane = threadIdx.x;    // lane == query row within this wave's 32
    const int  r    = lane & 15;      // WMMA column role
    const int  hh   = lane >> 4;
    const long g    = (long)blockIdx.x * MROW + lane;
    const int  b    = (int)(g / Q);
    const float cy  = coord[g * 2 + 0];
    const float cx  = coord[g * 2 + 1];
    const float cellh = cell[g * 2 + 0] * (float)H;
    const float cellw = cell[g * 2 + 1] * (float)W;

    // ---- per-row: all 4 shift areas (-> ensemble weights) + bilinear skip ----
    {
        float areas[4];
#pragma unroll
        for (int s = 0; s < 4; ++s) {
            float vx = (s & 2) ? 1.f : -1.f;
            float vy = (s & 1) ? 1.f : -1.f;
            float c0 = fminf(fmaxf(cy + vx * (1.f / H) + 1e-6f, -1.f + 1e-6f), 1.f - 1e-6f);
            float c1 = fminf(fmaxf(cx + vy * (1.f / W) + 1e-6f, -1.f + 1e-6f), 1.f - 1e-6f);
            int iy = (int)floorf((c0 + 1.f) * H * 0.5f); iy = iy < 0 ? 0 : (iy > H - 1 ? H - 1 : iy);
            int ix = (int)floorf((c1 + 1.f) * W * 0.5f); ix = ix < 0 ? 0 : (ix > W - 1 ? W - 1 : ix);
            float q0 = ((float)iy + 0.5f) * (2.f / H) - 1.f;
            float q1 = ((float)ix + 0.5f) * (2.f / W) - 1.f;
            areas[s] = fabsf((cy - q0) * H * (cx - q1) * W) + 1e-9f;
        }
        float tot = areas[0] + areas[1] + areas[2] + areas[3];
#pragma unroll
        for (int s = 0; s < 4; ++s) wgt_s[s][lane] = areas[3 - s] / tot;  // diagonal swap

        float yy = fminf(fmaxf((cy + 1.f) * H * 0.5f - 0.5f, 0.f), (float)(H - 1));
        float xx = fminf(fmaxf((cx + 1.f) * W * 0.5f - 0.5f, 0.f), (float)(W - 1));
        float y0f = floorf(yy), x0f = floorf(xx);
        float wy = yy - y0f, wx = xx - x0f;
        int y0 = (int)y0f, x0 = (int)x0f;
        int y1 = y0 + 1 > H - 1 ? H - 1 : y0 + 1;
        int x1 = x0 + 1 > W - 1 ? W - 1 : x0 + 1;
#pragma unroll
        for (int c = 0; c < 3; ++c) {
            const float* im = inp + ((size_t)(b * 3 + c)) * H * W;
            float v00 = im[y0 * W + x0], v01 = im[y0 * W + x1];
            float v10 = im[y1 * W + x0], v11 = im[y1 * W + x1];
            bil_s[lane][c] = v00 * (1 - wy) * (1 - wx) + v01 * (1 - wy) * wx +
                             v10 * wy * (1 - wx) + v11 * wy * wx;
        }
    }
    __syncthreads();

    v8f outAcc[2];
    outAcc[0] = (v8f){0, 0, 0, 0, 0, 0, 0, 0};
    outAcc[1] = (v8f){0, 0, 0, 0, 0, 0, 0, 0};

    for (int s = 0; s < 4; ++s) {
        // ---- shift geometry (each lane: its own query row) ----
        float vx = (s & 2) ? 1.f : -1.f;
        float vy = (s & 1) ? 1.f : -1.f;
        float c0 = fminf(fmaxf(cy + vx * (1.f / H) + 1e-6f, -1.f + 1e-6f), 1.f - 1e-6f);
        float c1 = fminf(fmaxf(cx + vy * (1.f / W) + 1e-6f, -1.f + 1e-6f), 1.f - 1e-6f);
        int iy = (int)floorf((c0 + 1.f) * H * 0.5f); iy = iy < 0 ? 0 : (iy > H - 1 ? H - 1 : iy);
        int ix = (int)floorf((c1 + 1.f) * W * 0.5f); ix = ix < 0 ? 0 : (ix > W - 1 ? W - 1 : ix);
        float q0 = ((float)iy + 0.5f) * (2.f / H) - 1.f;
        float q1 = ((float)ix + 0.5f) * (2.f / W) - 1.f;
        float rel0 = (cy - q0) * H, rel1 = (cx - q1) * W;

        // ---- build layer-0 panel row `lane`: q_feat (3x3 unfold) cols 0..575 ----
        const float* fb = feat + (size_t)b * CH * H * W;
        for (int c = 0; c < CH; ++c) {
            const float* fc = fb + (size_t)c * H * W;
#pragma unroll
            for (int t = 0; t < 9; ++t) {
                int ky = iy - 1 + t / 3, kx = ix - 1 + t % 3;
                float v = (ky >= 0 && ky < H && kx >= 0 && kx < W) ? fc[ky * W + kx] : 0.f;
                xpanel[lane * K0P + c * 9 + t] = f2bf(v);
            }
        }
        // ---- tail cols 576..639: rel/q/cell/coord + NeRF PE + zero pad ----
        {
            float tail[64];
#pragma unroll
            for (int i = 0; i < 64; ++i) tail[i] = 0.f;
            tail[0] = rel0;  tail[1] = rel1;
            tail[2] = q0;    tail[3] = q1;
            tail[4] = cellh; tail[5] = cellw;
            tail[6] = cy;    tail[7] = cx;
            float fr = 1.f;
#pragma unroll
            for (int f = 0; f < 10; ++f) {
                tail[8 + f * 4 + 0] = sinf(cy * fr);
                tail[8 + f * 4 + 1] = sinf(cx * fr);
                tail[8 + f * 4 + 2] = cosf(cy * fr);
                tail[8 + f * 4 + 3] = cosf(cx * fr);
                fr *= 2.f;
            }
#pragma unroll
            for (int i = 0; i < 64; ++i) xpanel[lane * K0P + 576 + i] = f2bf(tail[i]);
        }
        __syncthreads();

        // ---- MLP: 624->256->256->256->256 (WMMA bf16, f32 acc) ----
        run_layer32((const u32*)xpanel, K0P / 2, KT0, p0, b0, hA, lane);   // x -> hA
        __syncthreads();                                                   // xpanel dead
        run_layer32((const u32*)hA, HID / 2, KTH, p1, b1, hB, lane);       // hA -> hB
        __syncthreads();
        run_layer32((const u32*)hB, HID / 2, KTH, p2, b2, hA, lane);       // hB -> hA
        __syncthreads();
        run_layer32((const u32*)hA, HID / 2, KTH, p3, b3, hB, lane);       // hA -> hB
        __syncthreads();

        // ---- final layer 256->3 (single N-tile, cols 3..15 are zero weights) ----
        v8f acc[2];
        acc[0] = (v8f){0, 0, 0, 0, 0, 0, 0, 0};
        acc[1] = (v8f){0, 0, 0, 0, 0, 0, 0, 0};
        for (int kt = 0; kt < KTH; ++kt) {
            v16bf a0 = load_a_frag((const u32*)hB, HID / 2, lane, kt * 32, 0);
            v16bf a1 = load_a_frag((const u32*)hB, HID / 2, lane, kt * 32, 16);
            v16bf bb = load_b_frag(p4, kt, lane);
            acc[0] = __builtin_amdgcn_wmma_f32_16x16x32_bf16(
                false, a0, false, bb, (short)0, acc[0], false, false);
            acc[1] = __builtin_amdgcn_wmma_f32_16x16x32_bf16(
                false, a1, false, bb, (short)0, acc[1], false, false);
        }
        float bi = (r < 3) ? b4[r] : 0.f;   // D-layout: col == lane&15
#pragma unroll
        for (int m = 0; m < 2; ++m)
#pragma unroll
            for (int v = 0; v < 8; ++v) {
                int row = m * 16 + v + 8 * hh;
                outAcc[m][v] += (acc[m][v] + bi) * wgt_s[s][row];
            }
        __syncthreads();                    // before next shift rewrites panels
    }

    // ---- store: lanes with col<3 own output channels; rows across VGPRs ----
    if (r < 3) {
#pragma unroll
        for (int m = 0; m < 2; ++m)
#pragma unroll
            for (int v = 0; v < 8; ++v) {
                int  row = m * 16 + v + 8 * hh;
                long gg  = (long)blockIdx.x * MROW + row;
                out[gg * 3 + r] = outAcc[m][v] + bil_s[row][r];
            }
    }
}

// ---------------------------------------------------------------------------
extern "C" void kernel_launch(void* const* d_in, const int* in_sizes, int n_in,
                              void* d_out, int out_size, void* d_ws, size_t ws_size,
                              hipStream_t stream) {
    const float* inp    = (const float*)d_in[0];
    const float* coord  = (const float*)d_in[1];
    const float* cell   = (const float*)d_in[2];
    const float* conv_w = (const float*)d_in[3];
    const float* conv_b = (const float*)d_in[4];
    const float* w0 = (const float*)d_in[5];  const float* b0 = (const float*)d_in[6];
    const float* w1 = (const float*)d_in[7];  const float* b1 = (const float*)d_in[8];
    const float* w2 = (const float*)d_in[9];  const float* b2 = (const float*)d_in[10];
    const float* w3 = (const float*)d_in[11]; const float* b3 = (const float*)d_in[12];
    const float* w4 = (const float*)d_in[13]; const float* b4 = (const float*)d_in[14];
    float* out = (float*)d_out;

    char* ws    = (char*)d_ws;
    float* feat = (float*)ws;
    u32* p0 = (u32*)(ws + FEAT_BYTES);
    u32* p1 = (u32*)(ws + FEAT_BYTES + P0_BYTES);
    u32* p2 = (u32*)(ws + FEAT_BYTES + P0_BYTES + PH_BYTES);
    u32* p3 = (u32*)(ws + FEAT_BYTES + P0_BYTES + 2 * PH_BYTES);
    u32* p4 = (u32*)(ws + FEAT_BYTES + P0_BYTES + 3 * PH_BYTES);

    // 1) encoder conv -> feat (f32, NCHW)
    {
        int total = BATCH * CH * H * W;
        liif_conv3x3_kernel<<<(total + 255) / 256, 256, 0, stream>>>(inp, conv_w, conv_b, feat);
    }
    // 2) pack weights into WMMA-B fragment layout (bf16)
    liif_pack_kernel<<<(16 * KT0 * 256) / 256, 256, 0, stream>>>(w0, p0, 624, 256, KT0, 16);
    liif_pack_kernel<<<(16 * KTH * 256) / 256, 256, 0, stream>>>(w1, p1, 256, 256, KTH, 16);
    liif_pack_kernel<<<(16 * KTH * 256) / 256, 256, 0, stream>>>(w2, p2, 256, 256, KTH, 16);
    liif_pack_kernel<<<(16 * KTH * 256) / 256, 256, 0, stream>>>(w3, p3, 256, 256, KTH, 16);
    liif_pack_kernel<<<(1 * KTH * 256) / 256, 256, 0, stream>>>(w4, p4, 256, 3, KTH, 1);

    // 3) fused gather + 5-layer WMMA MLP + local ensemble + bilinear skip
    liif_fused_kernel<<<TILES, 32, 0, stream>>>(
        feat, inp, coord, cell,
        (const v4u*)p0, (const v4u*)p1, (const v4u*)p2, (const v4u*)p3, (const v4u*)p4,
        b0, b1, b2, b3, b4, out);
}
